// BaselineAttnDecoder_31482110279802
// MI455X (gfx1250) — matrix operand order repaired and checked by reading, not compiled, not run on hardware
//
#include <hip/hip_runtime.h>
#include <stdint.h>

// ---------------- problem sizes ----------------
#define BB      1280          // batch (128 imgs * 10 rounds)
#define LL      20            // question length
#define HH      512           // hidden
#define H3      1536          // 3*H
#define DD      300           // embed dim
#define DPAD    320           // 300 padded to mult of 32
#define KPAD    64            // attention key dim 50 padded
#define DECK    1324          // decoder GRU input (300 + 1024)
#define DECKPAD 1344
#define VV      8835          // vocab
#define VPAD    8864
#define IMGROWS 2048          // 128 imgs * 16 locations
#define IMGC    256
#define MAXLEN  21
#define OUTLD   (MAXLEN*DD)   // 6300: row stride of d_out per batch element

typedef __attribute__((ext_vector_type(16))) __bf16 v16bf;
typedef __attribute__((ext_vector_type(8)))  __bf16 v8bf;
typedef __attribute__((ext_vector_type(8)))  float  v8f;

union U16 { v16bf v; v8bf h[2]; };

// ---------------- WMMA GEMM:  C[M,N] = A[M,K] * W[N,K]^T + bias ----------------
// A, W bf16 row-major (K contiguous, zero-padded to K mult 32, N rows padded to mult 32).
// One wave -> 32x32 output tile (4x v_wmma_f32_16x16x32_bf16 per K-step).
__global__ void __launch_bounds__(128)
k_gemm(const __bf16* __restrict__ A, int lda,
       const __bf16* __restrict__ Bw, int ldb,
       const float* __restrict__ bias,
       float* __restrict__ C, int ldc,
       int M, int Npad, int K, int Nvalid)
{
    const int lane = threadIdx.x;                  // 0..31
    const int gw   = blockIdx.x * 4 + threadIdx.y; // global wave id
    const int ntiles = Npad >> 5;
    const int mtiles = M >> 5;
    if (gw >= ntiles * mtiles) return;
    const int mt = gw / ntiles;
    const int nt = gw - mt * ntiles;
    const int m0 = mt << 5, n0 = nt << 5;
    const int half = lane >> 4, r = lane & 15;

    const __bf16* A0 = A  + (size_t)(m0 + r) * lda;
    const __bf16* A1 = A0 + (size_t)16 * lda;
    const __bf16* B0 = Bw + (size_t)(n0 + r) * ldb;
    const __bf16* B1 = B0 + (size_t)16 * ldb;

    v8f acc00 = {}, acc01 = {}, acc10 = {}, acc11 = {};
    const int ka = 8 * half;    // A: VGPR0-3 -> K=8*half+0..7, VGPR4-7 -> +16
    const int kb = 8 * half;    // B: VGPR0-3 -> K=8*half+0..7, VGPR4-7 -> +16
    for (int k = 0; k < K; k += 32) {
        U16 a0, a1, b0, b1;
        a0.h[0] = *(const v8bf*)(A0 + k + ka);
        a0.h[1] = *(const v8bf*)(A0 + k + ka + 16);
        a1.h[0] = *(const v8bf*)(A1 + k + ka);
        a1.h[1] = *(const v8bf*)(A1 + k + ka + 16);
        b0.h[0] = *(const v8bf*)(B0 + k + kb);
        b0.h[1] = *(const v8bf*)(B0 + k + kb + 16);
        b1.h[0] = *(const v8bf*)(B1 + k + kb);
        b1.h[1] = *(const v8bf*)(B1 + k + kb + 16);
        acc00 = __builtin_amdgcn_wmma_f32_16x16x32_bf16(false, a0.v, false, b0.v, (short)0, acc00, false, false);
        acc01 = __builtin_amdgcn_wmma_f32_16x16x32_bf16(false, a0.v, false, b1.v, (short)0, acc01, false, false);
        acc10 = __builtin_amdgcn_wmma_f32_16x16x32_bf16(false, a1.v, false, b0.v, (short)0, acc10, false, false);
        acc11 = __builtin_amdgcn_wmma_f32_16x16x32_bf16(false, a1.v, false, b1.v, (short)0, acc11, false, false);
    }
    const int nA = n0 + r, nB = nA + 16;
    float bA = 0.f, bB = 0.f;
    if (bias) { if (nA < Nvalid) bA = bias[nA]; if (nB < Nvalid) bB = bias[nB]; }
#pragma unroll
    for (int j = 0; j < 8; ++j) {
        const int mA = m0 + 8 * half + j;
        const int mB = mA + 16;
        if (nA < Nvalid) {
            C[(size_t)mA * ldc + nA] = acc00[j] + bA;
            C[(size_t)mB * ldc + nA] = acc10[j] + bA;
        }
        if (nB < Nvalid) {
            C[(size_t)mA * ldc + nB] = acc01[j] + bB;
            C[(size_t)mB * ldc + nB] = acc11[j] + bB;
        }
    }
}

// ---------------- helpers ----------------
__global__ void k_zero_w(uint32_t* __restrict__ p, int n) {
    int i = blockIdx.x * 256 + threadIdx.x;
    if (i < n) p[i] = 0u;
}

// convert f32 [rows][cols] -> bf16 [rpad][cpad] zero-padded
__global__ void k_cvt_pad(const float* __restrict__ src, __bf16* __restrict__ dst,
                          int rows, int cols, int cpad, int total) {
    int i = blockIdx.x * 256 + threadIdx.x;
    if (i >= total) return;
    int r0 = i / cpad, c = i - r0 * cpad;
    float v = (r0 < rows && c < cols) ? src[(size_t)r0 * cols + c] : 0.f;
    dst[i] = (__bf16)v;
}

// embedding gather -> bf16, dst[b*dstride + c], c in [0,cwrite)
__global__ void k_gather(const int* __restrict__ toks, int tstride, int toff,
                         const float* __restrict__ emb, __bf16* __restrict__ dst,
                         int B, int cols, int cwrite, int dstride) {
    int i = blockIdx.x * 256 + threadIdx.x;
    int b = i / cwrite, c = i - b * cwrite;
    if (b >= B) return;
    int tok = toks[(size_t)b * tstride + toff];
    float v = (c < cols) ? emb[(size_t)tok * cols + c] : 0.f;
    dst[(size_t)b * dstride + c] = (__bf16)v;
}

__device__ __forceinline__ float sigm(float x) { return 1.f / (1.f + __expf(-x)); }

// GRU gates: gi/gh [B][1536] -> h[B][512] f32 + bf16 copy (+ masked enc output OR hc slot)
__global__ void k_gru(const float* __restrict__ gi, const float* __restrict__ gh,
                      float* __restrict__ h, __bf16* __restrict__ hb,
                      __bf16* __restrict__ extra, const int* __restrict__ lens,
                      int t, int mode) {
    int i = blockIdx.x * 256 + threadIdx.x;
    if (i >= BB * HH) return;
    int b = i >> 9, j = i & (HH - 1);
    const float* gib = gi + (size_t)b * H3;
    const float* ghb = gh + (size_t)b * H3;
    float rr = sigm(gib[j] + ghb[j]);
    float zz = sigm(gib[HH + j] + ghb[HH + j]);
    float nn = tanhf(gib[2 * HH + j] + rr * ghb[2 * HH + j]);
    float hn = (1.f - zz) * nn + zz * h[i];
    h[i] = hn;
    hb[i] = (__bf16)hn;
    if (mode == 0) {  // encoder: masked output into qh[B*L][H]
        float m = (t < lens[b]) ? hn : 0.f;
        extra[((size_t)b * LL + t) * HH + j] = (__bf16)m;
    } else {          // decoder: h part of [h|ctx] concat
        extra[(size_t)b * H3 + j] = (__bf16)hn;
    }
}

// attention scores + softmax (question: masked L=20; image: 16)
__global__ void k_attn(const float* __restrict__ qkey, const float* __restrict__ ikey,
                       const float* __restrict__ av, const int* __restrict__ lens,
                       float* __restrict__ qw, float* __restrict__ iw) {
    int b = blockIdx.x * 64 + threadIdx.x;
    if (b >= BB) return;
    float a[50];
#pragma unroll
    for (int k = 0; k < 50; ++k) a[k] = av[(size_t)b * KPAD + k];
    int len = lens[b];
    float e[LL], mx = -1e30f;
    for (int l = 0; l < LL; ++l) {
        if (l < len) {
            const float* qk = qkey + ((size_t)b * LL + l) * KPAD;
            float s = 0.f;
#pragma unroll
            for (int k = 0; k < 50; ++k) s += qk[k] * a[k];
            e[l] = s; mx = fmaxf(mx, s);
        } else e[l] = -1e30f;
    }
    float sum = 0.f;
    for (int l = 0; l < LL; ++l) {
        float ex = (l < len) ? __expf(e[l] - mx) : 0.f;
        e[l] = ex; sum += ex;
    }
    float inv = 1.f / sum;
    for (int l = 0; l < LL; ++l) qw[(size_t)b * LL + l] = e[l] * inv;

    int ib = (b / 10) * 16;
    float ei[16]; mx = -1e30f;
    for (int l = 0; l < 16; ++l) {
        const float* ik = ikey + (size_t)(ib + l) * KPAD;
        float s = 0.f;
#pragma unroll
        for (int k = 0; k < 50; ++k) s += ik[k] * a[k];
        ei[l] = s; mx = fmaxf(mx, s);
    }
    sum = 0.f;
    for (int l = 0; l < 16; ++l) { ei[l] = __expf(ei[l] - mx); sum += ei[l]; }
    float invi = 1.f / sum;
    for (int l = 0; l < 16; ++l) iw[(size_t)b * 16 + l] = ei[l] * invi;
}

// ctx = [qc|ic] (1024) -> bf16 into hc[:,512:1536] and xdec[:,300:1324]
__global__ void k_ctx(const float* __restrict__ qw, const float* __restrict__ iw,
                      const float* __restrict__ qval, const float* __restrict__ ival,
                      __bf16* __restrict__ hc, __bf16* __restrict__ xdec) {
    int i = blockIdx.x * 256 + threadIdx.x;
    if (i >= BB * 1024) return;
    int b = i >> 10, j = i & 1023;
    float v = 0.f;
    if (j < HH) {
        for (int l = 0; l < LL; ++l)
            v += qw[(size_t)b * LL + l] * qval[((size_t)b * LL + l) * HH + j];
    } else {
        int jj = j - HH;
        int ib = (b / 10) * 16;
        for (int l = 0; l < 16; ++l)
            v += iw[(size_t)b * 16 + l] * ival[(size_t)(ib + l) * HH + jj];
    }
    hc[(size_t)b * H3 + HH + j]        = (__bf16)v;
    xdec[(size_t)b * DECKPAD + DD + j] = (__bf16)v;
}

// step-19 output -> bf16 padded (for vocab GEMM)
__global__ void k_cvt_o(const float* __restrict__ out, __bf16* __restrict__ ob) {
    int i = blockIdx.x * 256 + threadIdx.x;
    if (i >= BB * DPAD) return;
    int b = i / DPAD, c = i - b * DPAD;
    float v = (c < DD) ? out[(size_t)b * OUTLD + 19 * DD + c] : 0.f;
    ob[i] = (__bf16)v;
}

__global__ void k_argmax(const float* __restrict__ logits, int* __restrict__ amax) {
    int b = blockIdx.x * 64 + threadIdx.x;
    if (b >= BB) return;
    const float* p = logits + (size_t)b * VPAD;
    float best = p[0]; int bi = 0;
    for (int v = 1; v < VV; ++v) {
        float x = p[v];
        if (x > best) { best = x; bi = v; }
    }
    amax[b] = bi;
}

// ---------------- host-side launch helpers ----------------
static inline void gemm(hipStream_t s, const __bf16* A, int lda, const __bf16* B, int ldb,
                        const float* bias, float* C, int ldc, int M, int Npad, int K, int Nvalid) {
    int tiles = (M / 32) * (Npad / 32);
    dim3 grid((tiles + 3) / 4), blk(32, 4);
    k_gemm<<<grid, blk, 0, s>>>(A, lda, B, ldb, bias, C, ldc, M, Npad, K, Nvalid);
}
static inline void cvt(hipStream_t s, const float* src, __bf16* dst,
                       int rows, int cols, int cpad, int rpad) {
    int n = rpad * cpad;
    k_cvt_pad<<<(n + 255) / 256, 256, 0, s>>>(src, dst, rows, cols, cpad, n);
}
static inline void zero_bytes(hipStream_t s, void* p, size_t bytes) {
    int n = (int)(bytes / 4);
    k_zero_w<<<(n + 255) / 256, 256, 0, s>>>((uint32_t*)p, n);
}

extern "C" void kernel_launch(void* const* d_in, const int* in_sizes, int n_in,
                              void* d_out, int out_size, void* d_ws, size_t ws_size,
                              hipStream_t stream)
{
    (void)in_sizes; (void)n_in; (void)out_size; (void)ws_size;
    const float* img   = (const float*)d_in[0];
    const int*   ques  = (const int*)  d_in[1];
    const int*   ans   = (const int*)  d_in[2];
    const int*   lens  = (const int*)  d_in[3];
    const float* embW  = (const float*)d_in[4];
    const float* eWih  = (const float*)d_in[5];
    const float* eWhh  = (const float*)d_in[6];
    const float* eBih  = (const float*)d_in[7];
    const float* eBhh  = (const float*)d_in[8];
    const float* dWih  = (const float*)d_in[9];
    const float* dWhh  = (const float*)d_in[10];
    const float* dBih  = (const float*)d_in[11];
    const float* dBhh  = (const float*)d_in[12];
    const float* qkW   = (const float*)d_in[13];
    const float* qkBv  = (const float*)d_in[14];
    const float* qvW   = (const float*)d_in[15];
    const float* qvBv  = (const float*)d_in[16];
    const float* ikW   = (const float*)d_in[17];
    const float* ikBv  = (const float*)d_in[18];
    const float* ivW   = (const float*)d_in[19];
    const float* ivBv  = (const float*)d_in[20];
    const float* akW   = (const float*)d_in[21];
    const float* akBv  = (const float*)d_in[22];
    const float* outW  = (const float*)d_in[23];
    const float* outBv = (const float*)d_in[24];
    const float* wdB   = (const float*)d_in[25];
    float* out = (float*)d_out;

    // ---- carve workspace ----
    char* base = (char*)d_ws;
    size_t off = 0;
    auto carve = [&](size_t bytes) -> void* {
        void* p = base + off;
        off = (off + bytes + 255) & ~(size_t)255;
        return p;
    };
    __bf16* wEih = (__bf16*)carve((size_t)H3 * DPAD * 2);
    __bf16* wEhh = (__bf16*)carve((size_t)H3 * HH * 2);
    __bf16* wDih = (__bf16*)carve((size_t)H3 * DECKPAD * 2);
    __bf16* wDhh = (__bf16*)carve((size_t)H3 * HH * 2);
    __bf16* wQk  = (__bf16*)carve((size_t)KPAD * HH * 2);
    __bf16* wQv  = (__bf16*)carve((size_t)HH * HH * 2);
    __bf16* wIk  = (__bf16*)carve((size_t)KPAD * IMGC * 2);
    __bf16* wIv  = (__bf16*)carve((size_t)HH * IMGC * 2);
    __bf16* wAk  = (__bf16*)carve((size_t)KPAD * HH * 2);
    __bf16* wOut = (__bf16*)carve((size_t)DPAD * H3 * 2);
    __bf16* wEmb = (__bf16*)carve((size_t)VPAD * DPAD * 2);
    __bf16* imgB = (__bf16*)carve((size_t)IMGROWS * IMGC * 2);
    __bf16* xenc = (__bf16*)carve((size_t)BB * DPAD * 2);
    __bf16* hb   = (__bf16*)carve((size_t)BB * HH * 2);
    __bf16* qh   = (__bf16*)carve((size_t)BB * LL * HH * 2);
    __bf16* xdec = (__bf16*)carve((size_t)BB * DECKPAD * 2);
    __bf16* hc   = (__bf16*)carve((size_t)BB * H3 * 2);
    __bf16* ob   = (__bf16*)carve((size_t)BB * DPAD * 2);
    float* hf    = (float*)carve((size_t)BB * HH * 4);
    float* gi    = (float*)carve((size_t)BB * H3 * 4);
    float* gh    = (float*)carve((size_t)BB * H3 * 4);
    float* qkey  = (float*)carve((size_t)BB * LL * KPAD * 4);
    float* qval  = (float*)carve((size_t)BB * LL * HH * 4);
    float* ikey  = (float*)carve((size_t)IMGROWS * KPAD * 4);
    float* ival  = (float*)carve((size_t)IMGROWS * HH * 4);
    float* af    = (float*)carve((size_t)BB * KPAD * 4);
    float* qwB   = (float*)carve((size_t)BB * LL * 4);
    float* iwB   = (float*)carve((size_t)BB * 16 * 4);
    float* logit = (float*)carve((size_t)BB * VPAD * 4);
    int*   amax  = (int*)  carve((size_t)BB * 4);

    // ---- setup: weight conversion (fp32 -> padded bf16) ----
    cvt(stream, eWih, wEih, H3, DD,   DPAD,    H3);
    cvt(stream, eWhh, wEhh, H3, HH,   HH,      H3);
    cvt(stream, dWih, wDih, H3, DECK, DECKPAD, H3);
    cvt(stream, dWhh, wDhh, H3, HH,   HH,      H3);
    cvt(stream, qkW,  wQk,  50, HH,   HH,      KPAD);
    cvt(stream, qvW,  wQv,  HH, HH,   HH,      HH);
    cvt(stream, ikW,  wIk,  50, IMGC, IMGC,    KPAD);
    cvt(stream, ivW,  wIv,  HH, IMGC, IMGC,    HH);
    cvt(stream, akW,  wAk,  50, HH,   HH,      KPAD);
    cvt(stream, outW, wOut, DD, H3,   H3,      DPAD);
    cvt(stream, embW, wEmb, VV, DD,   DPAD,    VPAD);
    cvt(stream, img,  imgB, IMGROWS, IMGC, IMGC, IMGROWS);

    // image key/value (computed at 128-image granularity; batch indexes with b/10)
    gemm(stream, imgB, IMGC, wIk, IMGC, ikBv, ikey, KPAD, IMGROWS, KPAD, IMGC, 50);
    gemm(stream, imgB, IMGC, wIv, IMGC, ivBv, ival, HH,   IMGROWS, HH,   IMGC, HH);

    zero_bytes(stream, hf,   (size_t)BB * HH * 4);
    zero_bytes(stream, hb,   (size_t)BB * HH * 2);
    zero_bytes(stream, xdec, (size_t)BB * DECKPAD * 2);   // keeps K-pad cols zero forever

    // ---- encoder GRU ----
    for (int t = 0; t < LL; ++t) {
        int n = BB * DPAD;
        k_gather<<<(n + 255) / 256, 256, 0, stream>>>(ques, LL, t, embW, xenc, BB, DD, DPAD, DPAD);
        gemm(stream, xenc, DPAD, wEih, DPAD, eBih, gi, H3, BB, H3, DPAD, H3);
        gemm(stream, hb,   HH,   wEhh, HH,   eBhh, gh, H3, BB, H3, HH,   H3);
        k_gru<<<(BB * HH + 255) / 256, 256, 0, stream>>>(gi, gh, hf, hb, qh, lens, t, 0);
    }

    // question key/value over all (b,l) rows
    gemm(stream, qh, HH, wQk, HH, qkBv, qkey, KPAD, BB * LL, KPAD, HH, 50);
    gemm(stream, qh, HH, wQv, HH, qvBv, qval, HH,   BB * LL, HH,   HH, HH);

    // ---- decoder ----
    zero_bytes(stream, hf, (size_t)BB * HH * 4);
    zero_bytes(stream, hb, (size_t)BB * HH * 2);

    for (int s = 0; s < MAXLEN; ++s) {
        // dec_in gather (ans token for s<20, argmax token at s==20)
        int n = BB * DD;
        if (s < MAXLEN - 1)
            k_gather<<<(n + 255) / 256, 256, 0, stream>>>(ans, MAXLEN, s, embW, xdec, BB, DD, DD, DECKPAD);
        else
            k_gather<<<(n + 255) / 256, 256, 0, stream>>>(amax, 1, 0, embW, xdec, BB, DD, DD, DECKPAD);

        // a = h @ ak^T + b ; attention ; ctx
        gemm(stream, hb, HH, wAk, HH, akBv, af, KPAD, BB, KPAD, HH, 50);
        k_attn<<<(BB + 63) / 64, 64, 0, stream>>>(qkey, ikey, af, lens, qwB, iwB);
        k_ctx<<<(BB * 1024 + 255) / 256, 256, 0, stream>>>(qwB, iwB, qval, ival, hc, xdec);

        // decoder GRU
        gemm(stream, xdec, DECKPAD, wDih, DECKPAD, dBih, gi, H3, BB, H3, DECKPAD, H3);
        gemm(stream, hb,   HH,      wDhh, HH,      dBhh, gh, H3, BB, H3, HH,      H3);
        k_gru<<<(BB * HH + 255) / 256, 256, 0, stream>>>(gi, gh, hf, hb, hc, lens, 0, 1);

        // o = [h|ctx] @ out^T + b  ->  d_out[:, s, :]
        gemm(stream, hc, H3, wOut, H3, outBv, out + (size_t)s * DD, OUTLD, BB, DPAD, H3, DD);

        if (s == MAXLEN - 2) {  // vocabulary projection + argmax for last dec_in
            int m = BB * DPAD;
            k_cvt_o<<<(m + 255) / 256, 256, 0, stream>>>(out, ob);
            gemm(stream, ob, DPAD, wEmb, DPAD, wdB, logit, VPAD, BB, VPAD, DPAD, VV);
            k_argmax<<<(BB + 63) / 64, 64, 0, stream>>>(logit, amax);
        }
    }
}
